// SAEEncoder_21809843929581
// MI455X (gfx1250) — matrix operand order repaired and checked by reading, not compiled, not run on hardware
//
#include <hip/hip_runtime.h>
#include <hip/hip_bf16.h>

// SAE forward for MI455X (gfx1250, wave32, WMMA).
// encode GEMM (bf16 wmma, f32 acc, BM=256 x BN=128 tiles, software-pipelined
// register->LDS staging) -> z ; radix-select top-64 threshold + mask ;
// decode GEMM (bf16 wmma) with per-(16-row,32-latent) sparsity skip flags.

#define HIDDEN 4096
#define LATENT 32768
#define KTOP   64
#define BATCH  4096

typedef __attribute__((ext_vector_type(16))) __bf16 v16bf;
typedef __attribute__((ext_vector_type(8)))  float  v8f;

#define LDS_STRIDE 40  // bf16 elems; 80B row stride => conflict-free ds_read_b128 across 16 lanes

union BfPack4 { __bf16 h[4]; uint2 u; };
union FragU   { v16bf v; uint4 u[2]; };

// A-matrix 16x32 bf16 fragment (ISA 7.12.2): lane<16 holds K0..7 & K16..23,
// lane>=16 holds K8..15 & K24..31 for row M = lane&15. Two 16B LDS reads.
__device__ __forceinline__ v16bf load_frag_a(const __bf16* rowp, int hi) {
  FragU f;
  f.u[0] = *(const uint4*)(rowp + hi * 8);
  f.u[1] = *(const uint4*)(rowp + 16 + hi * 8);
  return f.v;
}
// B-matrix 32x16 bf16 fragment: lane<16 holds K0..7 & K16..23, lane>=16 holds K8..15 & K24..31
// for col N = lane&15 (tile stored [N][K] row-major in LDS). Two 16B reads.
__device__ __forceinline__ v16bf load_frag_b(const __bf16* rowp, int hi) {
  FragU f;
  f.u[0] = *(const uint4*)(rowp + hi * 8);
  f.u[1] = *(const uint4*)(rowp + 16 + hi * 8);
  return f.v;
}

// Fetch this thread's share of a ROWSx32 f32 tile (row-major, leading dim LD)
// into registers (ROWS/32 x float4).
template<int LD, int ROWS>
__device__ __forceinline__ void global_fetch(const float* __restrict__ src,
                                             int row0, int k0, int tid,
                                             float4* __restrict__ r4) {
#pragma unroll
  for (int j = 0; j < ROWS / 32; ++j) {
    const int r = (tid >> 3) + 32 * j;  // 32 rows per sweep
    const int c = (tid & 7) * 4;        // 8 float4 chunks per 32-wide row
    r4[j] = *(const float4*)(src + (size_t)(row0 + r) * LD + k0 + c);
  }
}

// Convert staged registers to bf16 and commit to LDS.
template<int ROWS>
__device__ __forceinline__ void lds_commit(const float4* __restrict__ r4,
                                           __bf16 (*dst)[LDS_STRIDE], int tid) {
#pragma unroll
  for (int j = 0; j < ROWS / 32; ++j) {
    const int r = (tid >> 3) + 32 * j;
    const int c = (tid & 7) * 4;
    BfPack4 p;
    p.h[0] = (__bf16)r4[j].x; p.h[1] = (__bf16)r4[j].y;
    p.h[2] = (__bf16)r4[j].z; p.h[3] = (__bf16)r4[j].w;
    *(uint2*)&dst[r][c] = p.u;          // 8B aligned LDS store
  }
}

#define WMMA_BF16(A, B, C) \
  __builtin_amdgcn_wmma_f32_16x16x32_bf16(false, (A), false, (B), (short)0, (C), false, false)

// ---------------- encoder: z = relu(x @ W_enc^T + b_enc) ----------------
__global__ __launch_bounds__(256)
void sae_encode_gemm(const float* __restrict__ x,      // [BATCH, HIDDEN]
                     const float* __restrict__ W_enc,  // [LATENT, HIDDEN]
                     const float* __restrict__ b_enc,  // [LATENT]
                     float* __restrict__ z)            // [BATCH, LATENT]
{
  __shared__ alignas(16) __bf16 sA[256][LDS_STRIDE];
  __shared__ alignas(16) __bf16 sB[128][LDS_STRIDE];
  const int tid  = threadIdx.x;
  const int lane = tid & 31;
  const int ms   = tid >> 5;   // wave id -> 32-row M strip (2 x 16)
  const int hi   = lane >> 4;
  const int lo   = lane & 15;
  const int m0 = blockIdx.x * 256;   // batch tile (fast dim -> W_enc tiles shared via L2)
  const int n0 = blockIdx.y * 128;   // latent tile

  v8f acc0[8] = {}, acc1[8] = {};
  float4 ra[8], rb[4];
  global_fetch<HIDDEN, 256>(x,     m0, 0, tid, ra);
  global_fetch<HIDDEN, 128>(W_enc, n0, 0, tid, rb);

  for (int k0 = 0; k0 < HIDDEN; k0 += 32) {
    lds_commit<256>(ra, sA, tid);
    lds_commit<128>(rb, sB, tid);
    __syncthreads();
    if (k0 + 32 < HIDDEN) {   // prefetch next K tile; latency hidden by wmma chain
      global_fetch<HIDDEN, 256>(x,     m0, k0 + 32, tid, ra);
      global_fetch<HIDDEN, 128>(W_enc, n0, k0 + 32, tid, rb);
    }
    const v16bf af0 = load_frag_a(&sA[ms * 32 + lo][0], hi);
    const v16bf af1 = load_frag_a(&sA[ms * 32 + 16 + lo][0], hi);
#pragma unroll
    for (int nt = 0; nt < 8; ++nt) {
      const v16bf bf = load_frag_b(&sB[nt * 16 + lo][0], hi);
      acc0[nt] = WMMA_BF16(af0, bf, acc0[nt]);
      acc1[nt] = WMMA_BF16(af1, bf, acc1[nt]);
    }
    __syncthreads();
  }
  // C layout: VGPR i -> M = i + 8*hi, N = lo
#pragma unroll
  for (int nt = 0; nt < 8; ++nt) {
    const int n = n0 + nt * 16 + lo;
    const float be = b_enc[n];
#pragma unroll
    for (int i = 0; i < 8; ++i) {
      const int mbase = m0 + ms * 32 + hi * 8 + i;
      float v0 = acc0[nt][i] + be;
      float v1 = acc1[nt][i] + be;
      z[(size_t)mbase * LATENT + n]        = v0 > 0.f ? v0 : 0.f;
      z[(size_t)(mbase + 16) * LATENT + n] = v1 > 0.f ? v1 : 0.f;
    }
  }
}

// ---------------- top-k threshold + mask + sparsity flags ----------------
__global__ void sae_zero_flags(unsigned int* __restrict__ flags, int n) {
  const int i = blockIdx.x * 256 + threadIdx.x;
  if (i < n) flags[i] = 0u;
}

__global__ __launch_bounds__(256)
void sae_topk_mask(float* __restrict__ z, unsigned int* __restrict__ flags,
                   int use_flags) {
  extern __shared__ float srow[];            // LATENT floats (128 KB dynamic LDS)
  __shared__ unsigned int hist[256];
  __shared__ unsigned int s_prefix, s_want;
  const int tid = threadIdx.x;
  const int b   = blockIdx.x;
  float* zrow = z + (size_t)b * LATENT;

  for (int i = tid; i < LATENT / 4; i += 256)
    ((float4*)srow)[i] = ((const float4*)zrow)[i];
  if (tid == 0) { s_prefix = 0u; s_want = KTOP; }
  __syncthreads();

  // 4-pass radix select for the KTOP-th largest. Post-ReLU values are all
  // non-negative -> IEEE bits order as unsigned ints.
  for (int shift = 24; shift >= 0; shift -= 8) {
    hist[tid] = 0u;                          // blockDim == 256
    __syncthreads();
    const unsigned int prefix = s_prefix;
    const unsigned int pmask  = (shift == 24) ? 0u : (0xFFFFFFFFu << (shift + 8));
    for (int i = tid; i < LATENT; i += 256) {
      const unsigned int u = __float_as_uint(srow[i]);
      if ((u & pmask) == prefix)
        atomicAdd(&hist[(u >> shift) & 255u], 1u);
    }
    __syncthreads();
    if (tid == 0) {
      unsigned int w = s_want, cum = 0u;
      int d = 255;
      for (; d > 0; --d) {
        const unsigned int c = hist[d];
        if (cum + c >= w) break;
        cum += c;
      }
      s_prefix = prefix | ((unsigned int)d << shift);
      s_want   = w - cum;
    }
    __syncthreads();
  }

  const float thr = __uint_as_float(s_prefix);   // exact 64th-largest value
  const int rowgrp = b >> 4;
  for (int i = tid; i < LATENT; i += 256) {
    const float v = srow[i];
    const float o = (v >= thr) ? v : 0.0f;       // ties kept, like reference
    zrow[i] = o;
    if (use_flags && o != 0.0f)
      atomicOr(&flags[(size_t)rowgrp * (LATENT / 32) + (i >> 5)], 1u);
  }
}

// ---------------- decoder: recon = z_sparse @ W_dec^T + b_dec ----------------
template<bool USE_FLAGS>
__global__ __launch_bounds__(256)
void sae_decode_gemm(const float* __restrict__ z,      // [BATCH, LATENT] (masked)
                     const float* __restrict__ W_dec,  // [HIDDEN, LATENT]
                     const float* __restrict__ b_dec,  // [HIDDEN]
                     float* __restrict__ recon,        // [BATCH, HIDDEN]
                     const unsigned int* __restrict__ flags) {
  __shared__ alignas(16) __bf16 sA[256][LDS_STRIDE];
  __shared__ alignas(16) __bf16 sB[128][LDS_STRIDE];
  const int tid  = threadIdx.x;
  const int lane = tid & 31;
  const int ms   = tid >> 5;
  const int hi   = lane >> 4;
  const int lo   = lane & 15;
  const int m0 = blockIdx.x * 256;
  const int n0 = blockIdx.y * 128;
  const int rg0 = (m0 >> 4) + ms * 2;  // this wave's two 16-batch-row groups
  const int rg1 = rg0 + 1;

  v8f acc0[8] = {}, acc1[8] = {};
  float4 ra[8], rb[4];
  global_fetch<LATENT, 256>(z,     m0, 0, tid, ra);
  global_fetch<LATENT, 128>(W_dec, n0, 0, tid, rb);

  for (int k0 = 0; k0 < LATENT; k0 += 32) {
    lds_commit<256>(ra, sA, tid);
    lds_commit<128>(rb, sB, tid);
    __syncthreads();
    if (k0 + 32 < LATENT) {
      global_fetch<LATENT, 256>(z,     m0, k0 + 32, tid, ra);
      global_fetch<LATENT, 128>(W_dec, n0, k0 + 32, tid, rb);
    }
    // Sparsity skip, wave-uniform per 16-row strip -> EXEC stays all-1s.
    bool a0 = true, a1 = true;
    if (USE_FLAGS) {
      const int ch = k0 >> 5;
      a0 = (flags[(size_t)rg0 * (LATENT / 32) + ch] != 0u);
      a1 = (flags[(size_t)rg1 * (LATENT / 32) + ch] != 0u);
    }
    if (a0 && a1) {
      const v16bf af0 = load_frag_a(&sA[ms * 32 + lo][0], hi);
      const v16bf af1 = load_frag_a(&sA[ms * 32 + 16 + lo][0], hi);
#pragma unroll
      for (int nt = 0; nt < 8; ++nt) {
        const v16bf bf = load_frag_b(&sB[nt * 16 + lo][0], hi);
        acc0[nt] = WMMA_BF16(af0, bf, acc0[nt]);
        acc1[nt] = WMMA_BF16(af1, bf, acc1[nt]);
      }
    } else if (a0) {
      const v16bf af0 = load_frag_a(&sA[ms * 32 + lo][0], hi);
#pragma unroll
      for (int nt = 0; nt < 8; ++nt) {
        const v16bf bf = load_frag_b(&sB[nt * 16 + lo][0], hi);
        acc0[nt] = WMMA_BF16(af0, bf, acc0[nt]);
      }
    } else if (a1) {
      const v16bf af1 = load_frag_a(&sA[ms * 32 + 16 + lo][0], hi);
#pragma unroll
      for (int nt = 0; nt < 8; ++nt) {
        const v16bf bf = load_frag_b(&sB[nt * 16 + lo][0], hi);
        acc1[nt] = WMMA_BF16(af1, bf, acc1[nt]);
      }
    }
    __syncthreads();
  }
#pragma unroll
  for (int nt = 0; nt < 8; ++nt) {
    const int n = n0 + nt * 16 + lo;
    const float bd = b_dec[n];
#pragma unroll
    for (int i = 0; i < 8; ++i) {
      const int mbase = m0 + ms * 32 + hi * 8 + i;
      recon[(size_t)mbase * HIDDEN + n]        = acc0[nt][i] + bd;
      recon[(size_t)(mbase + 16) * HIDDEN + n] = acc1[nt][i] + bd;
    }
  }
}

// ---------------- host entry ----------------
extern "C" void kernel_launch(void* const* d_in, const int* in_sizes, int n_in,
                              void* d_out, int out_size, void* d_ws, size_t ws_size,
                              hipStream_t stream) {
  (void)in_sizes; (void)n_in; (void)out_size;
  const float* x     = (const float*)d_in[0];
  const float* W_enc = (const float*)d_in[1];
  const float* b_enc = (const float*)d_in[2];
  const float* W_dec = (const float*)d_in[3];
  const float* b_dec = (const float*)d_in[4];
  float* recon = (float*)d_out;                                // [BATCH, HIDDEN]
  float* z     = (float*)d_out + (size_t)BATCH * HIDDEN;       // [BATCH, LATENT]

  const int nflags = (BATCH / 16) * (LATENT / 32);             // 262144 words (1 MB)
  const bool use_flags = ws_size >= (size_t)nflags * sizeof(unsigned int);
  unsigned int* flags = (unsigned int*)d_ws;

  if (use_flags)
    sae_zero_flags<<<dim3((nflags + 255) / 256), dim3(256), 0, stream>>>(flags, nflags);

  sae_encode_gemm<<<dim3(BATCH / 256, LATENT / 128), dim3(256), 0, stream>>>(
      x, W_enc, b_enc, z);

  sae_topk_mask<<<dim3(BATCH), dim3(256), (size_t)LATENT * sizeof(float), stream>>>(
      z, use_flags ? flags : (unsigned int*)nullptr, use_flags ? 1 : 0);

  if (use_flags)
    sae_decode_gemm<true><<<dim3(BATCH / 256, HIDDEN / 128), dim3(256), 0, stream>>>(
        z, W_dec, b_dec, recon, flags);
  else
    sae_decode_gemm<false><<<dim3(BATCH / 256, HIDDEN / 128), dim3(256), 0, stream>>>(
        z, W_dec, b_dec, recon, nullptr);
}
